// MultiheadAttention_7782480740391
// MI455X (gfx1250) — matrix operand order repaired and checked by reading, not compiled, not run on hardware
//
#include <hip/hip_runtime.h>
#include <hip/hip_bf16.h>
#include <cstdint>

typedef __attribute__((ext_vector_type(16))) _Float16 v16h;
typedef __attribute__((ext_vector_type(8)))  _Float16 v8h;
typedef __attribute__((ext_vector_type(4)))  _Float16 v4h;
typedef __attribute__((ext_vector_type(8)))  float    v8f;

constexpr int BB  = 2;
constexpr int SS  = 2048;
constexpr int DD  = 768;
constexpr int HH  = 12;
constexpr int DHH = 64;
constexpr int M_ROWS = BB * SS;   // 4096
constexpr int N_QKV  = 3 * DD;    // 2304

__device__ __forceinline__ v8f wmma32(v16h a, v16h b, v8f c) {
  // v_wmma_f32_16x16x32_f16: D = A(16x32 f16) * B(32x16 f16) + C(16x16 f32)
  return __builtin_amdgcn_wmma_f32_16x16x32_f16(false, a, false, b, (short)0, c,
                                                false, false);
}

// A-matrix 16x32 f16 fragment (ISA 7.12.2): lane holds row m0+(lane&15);
// lanes 0-15: K = k0+[0..7], k0+[16..23]; lanes 16-31: +8.
__device__ __forceinline__ v16h frag_a_f16(const _Float16* __restrict__ p, int ld,
                                           int m0, int k0) {
  const int lane = threadIdx.x & 31;
  const _Float16* r =
      p + (size_t)(m0 + (lane & 15)) * ld + k0 + ((lane < 16) ? 0 : 8);
  v8h lo = *(const v8h*)r;
  v8h hi = *(const v8h*)(r + 16);
  return __builtin_shufflevector(lo, hi, 0, 1, 2, 3, 4, 5, 6, 7, 8, 9, 10, 11,
                                 12, 13, 14, 15);
}

// B-matrix 32x16 f16 fragment: lane holds column n0+(lane&15);
// lanes 0-15: K = k0+[0..7], k0+[16..23]; lanes 16-31: +8.
// Source is row n of a row-major [N,K] array (== column n of B).
__device__ __forceinline__ v16h frag_b_f16(const _Float16* __restrict__ p, int ld,
                                           int n0, int k0) {
  const int lane = threadIdx.x & 31;
  const _Float16* c =
      p + (size_t)(n0 + (lane & 15)) * ld + k0 + ((lane < 16) ? 0 : 8);
  v8h lo = *(const v8h*)c;
  v8h hi = *(const v8h*)(c + 16);
  return __builtin_shufflevector(lo, hi, 0, 1, 2, 3, 4, 5, 6, 7, 8, 9, 10, 11,
                                 12, 13, 14, 15);
}

// ---------------------------------------------------------------- fp32 -> f16
__global__ __launch_bounds__(256) void cvt_f32_to_f16_kernel(
    const float* __restrict__ src, _Float16* __restrict__ dst, int n4) {
  int i = blockIdx.x * 256 + threadIdx.x;
  if (i < n4) {
    const float4 v = ((const float4*)src)[i];
    v4h o;
    o[0] = (_Float16)v.x; o[1] = (_Float16)v.y;
    o[2] = (_Float16)v.z; o[3] = (_Float16)v.w;
    ((v4h*)dst)[i] = o;
  }
}

// ------------------------------------------------------------------ QKV GEMM
// C[m,n] = sum_k X[m,k] * Wqkv[n,k] + b[n];  M=4096, N=2304, K=768
// 32x64 output tile per wave: 2 A-frags x 4 B-frags -> 8 WMMAs per K-step.
// Q,K stored [B,H,S,dh]; V stored transposed [B,H,dh,S] for the PV stage.
__global__ __launch_bounds__(256) void qkv_gemm_kernel(
    const _Float16* __restrict__ x, const _Float16* __restrict__ w,
    const float* __restrict__ bias, _Float16* __restrict__ q_ws,
    _Float16* __restrict__ k_ws, _Float16* __restrict__ v_ws) {
  const int wave = threadIdx.x >> 5, lane = threadIdx.x & 31;
  const int tile = blockIdx.x * 8 + wave;
  const int m0 = (tile % (M_ROWS / 32)) * 32;
  const int n0 = (tile / (M_ROWS / 32)) * 64;

  v8f acc[2][4] = {};
  for (int k0 = 0; k0 < DD; k0 += 32) {
    const v16h a0 = frag_a_f16(x, DD, m0, k0);
    const v16h a1 = frag_a_f16(x, DD, m0 + 16, k0);
#pragma unroll
    for (int j = 0; j < 4; ++j) {
      const v16h bj = frag_b_f16(w, DD, n0 + j * 16, k0);
      acc[0][j] = wmma32(a0, bj, acc[0][j]);
      acc[1][j] = wmma32(a1, bj, acc[1][j]);
    }
  }

  const int half8 = (lane >> 4) << 3;
#pragma unroll
  for (int j = 0; j < 4; ++j) {
    const int n = n0 + j * 16 + (lane & 15);
    const float bn = bias[n];
    const int which = n / DD, d = n % DD, h = d / DHH, dh = d % DHH;
#pragma unroll
    for (int i = 0; i < 2; ++i) {
#pragma unroll
      for (int r = 0; r < 8; ++r) {
        const int m = m0 + i * 16 + r + half8;
        const int b = m / SS, s = m % SS;
        const _Float16 val = (_Float16)(acc[i][j][r] + bn);
        const size_t bh = (size_t)(b * HH + h);
        if (which == 0)
          q_ws[(bh * SS + s) * DHH + dh] = val;
        else if (which == 1)
          k_ws[(bh * SS + s) * DHH + dh] = val;
        else
          v_ws[(bh * DHH + dh) * SS + s] = val;   // V^T: [dh, s]
      }
    }
  }
}

// per-16-row-tile online-softmax update over a 32-key block (C-layout scores)
__device__ __forceinline__ void softmax_update(
    v8f& sc0, v8f& sc1, float* mrun, float* lrun, v8f& o0, v8f& o1, v8f& o2,
    v8f& o3, int q_base, int half8, int key0, int key1, bool pad0, bool pad1,
    _Float16* sp_tile, int n) {
  const float NEG_INF = -__builtin_inff();
#pragma unroll
  for (int r = 0; r < 8; ++r) {
    const int q = q_base + r + half8;
    float s0 = sc0[r] * 0.125f;               // 1/sqrt(64)
    float s1 = sc1[r] * 0.125f;
    if (pad0 || key0 > q) s0 = NEG_INF;
    if (pad1 || key1 > q) s1 = NEG_INF;
    float rm = fmaxf(s0, s1);
#pragma unroll
    for (int mk = 1; mk <= 8; mk <<= 1) rm = fmaxf(rm, __shfl_xor(rm, mk, 32));
    const float mnew = fmaxf(mrun[r], rm);
    const float alpha = (mnew == NEG_INF) ? 1.f : __expf(mrun[r] - mnew);
    const float p0 = (s0 == NEG_INF) ? 0.f : __expf(s0 - mnew);
    const float p1 = (s1 == NEG_INF) ? 0.f : __expf(s1 - mnew);
    float rs = p0 + p1;
#pragma unroll
    for (int mk = 1; mk <= 8; mk <<= 1) rs += __shfl_xor(rs, mk, 32);
    lrun[r] = lrun[r] * alpha + rs;
    mrun[r] = mnew;
    o0[r] *= alpha; o1[r] *= alpha; o2[r] *= alpha; o3[r] *= alpha;
    const int row = r + half8;
    sp_tile[row * 32 + n] = (_Float16)p0;
    sp_tile[row * 32 + 16 + n] = (_Float16)p1;
  }
}

// -------------------------------------------------------- flash attention
// One wave per (b, h, 32-row query block): two 16-row tiles share every
// K and V fragment. 32-key blocks, online softmax, P via per-wave LDS tile.
__global__ __launch_bounds__(128) void attn_kernel(
    const _Float16* __restrict__ q_ws, const _Float16* __restrict__ k_ws,
    const _Float16* __restrict__ v_ws, const unsigned char* __restrict__ pad,
    _Float16* __restrict__ o_ws) {
  __shared__ _Float16 sP[4][32 * 32];   // per-wave P tile (C->A layout bounce)
  const int wave = threadIdx.x >> 5, lane = threadIdx.x & 31;
  const int job = blockIdx.x * 4 + wave;
  const int qt = job & (SS / 32 - 1);   // 0..63
  const int bh = job >> 6;              // 0..23
  const int b = bh / HH, h = bh % HH;
  const int q0 = qt * 32;
  const _Float16* Qp = q_ws + (size_t)bh * SS * DHH;
  const _Float16* Kp = k_ws + (size_t)bh * SS * DHH;
  const _Float16* Vt = v_ws + (size_t)bh * DHH * SS;
  _Float16* sp = sP[wave];

  const v16h qa00 = frag_a_f16(Qp, DHH, q0, 0);        // rows q0..+15, dh 0..31
  const v16h qa01 = frag_a_f16(Qp, DHH, q0, 32);       // rows q0..+15, dh 32..63
  const v16h qa10 = frag_a_f16(Qp, DHH, q0 + 16, 0);   // rows q0+16..+31
  const v16h qa11 = frag_a_f16(Qp, DHH, q0 + 16, 32);

  const float NEG_INF = -__builtin_inff();
  float mrun[2][8], lrun[2][8];
#pragma unroll
  for (int t = 0; t < 2; ++t)
#pragma unroll
    for (int r = 0; r < 8; ++r) { mrun[t][r] = NEG_INF; lrun[t][r] = 0.f; }
  v8f o00 = {}, o01 = {}, o02 = {}, o03 = {};
  v8f o10 = {}, o11 = {}, o12 = {}, o13 = {};

  const int n = lane & 15;
  const int half8 = (lane >> 4) << 3;
  const int nblocks = qt + 1;           // causal: keys <= q0+31
  for (int kb = 0; kb < nblocks; ++kb) {
    const int kk = kb * 32;
    // S = Q @ K^T  (M=32 q rows, N=32 keys, K=64 dh); KV frags shared by tiles
    v8f sc00 = {}, sc01 = {}, sc10 = {}, sc11 = {};
    {
      const v16h bk00 = frag_b_f16(Kp, DHH, kk, 0);
      const v16h bk01 = frag_b_f16(Kp, DHH, kk, 32);
      sc00 = wmma32(qa00, bk00, sc00); sc00 = wmma32(qa01, bk01, sc00);
      sc10 = wmma32(qa10, bk00, sc10); sc10 = wmma32(qa11, bk01, sc10);
      const v16h bk10 = frag_b_f16(Kp, DHH, kk + 16, 0);
      const v16h bk11 = frag_b_f16(Kp, DHH, kk + 16, 32);
      sc01 = wmma32(qa00, bk10, sc01); sc01 = wmma32(qa01, bk11, sc01);
      sc11 = wmma32(qa10, bk10, sc11); sc11 = wmma32(qa11, bk11, sc11);
    }

    const int key0 = kk + n, key1 = kk + 16 + n;
    const bool pad0 = pad[b * SS + key0] != 0;
    const bool pad1 = pad[b * SS + key1] != 0;

    softmax_update(sc00, sc01, mrun[0], lrun[0], o00, o01, o02, o03, q0, half8,
                   key0, key1, pad0, pad1, sp, n);
    softmax_update(sc10, sc11, mrun[1], lrun[1], o10, o11, o12, o13, q0 + 16,
                   half8, key0, key1, pad0, pad1, sp + 16 * 32, n);
    asm volatile("s_wait_dscnt 0" ::: "memory");   // same-wave LDS RAW fence

    // reload P as two A-matrix 16x32 fragments
    const int ro = (lane & 15) * 32 + ((lane < 16) ? 0 : 8);
    const _Float16* pr0 = sp + ro;
    const _Float16* pr1 = sp + 16 * 32 + ro;
    v16h pa0 = __builtin_shufflevector(*(const v8h*)pr0, *(const v8h*)(pr0 + 16),
                                       0, 1, 2, 3, 4, 5, 6, 7, 8, 9, 10, 11, 12,
                                       13, 14, 15);
    v16h pa1 = __builtin_shufflevector(*(const v8h*)pr1, *(const v8h*)(pr1 + 16),
                                       0, 1, 2, 3, 4, 5, 6, 7, 8, 9, 10, 11, 12,
                                       13, 14, 15);
    // O += P @ V  (M=32, N=64 dh, K=32 keys); B[k,n]=V[k,n]=Vt[n,k]
    {
      const v16h bv0 = frag_b_f16(Vt, SS, 0, kk);
      o00 = wmma32(pa0, bv0, o00); o10 = wmma32(pa1, bv0, o10);
      const v16h bv1 = frag_b_f16(Vt, SS, 16, kk);
      o01 = wmma32(pa0, bv1, o01); o11 = wmma32(pa1, bv1, o11);
      const v16h bv2 = frag_b_f16(Vt, SS, 32, kk);
      o02 = wmma32(pa0, bv2, o02); o12 = wmma32(pa1, bv2, o12);
      const v16h bv3 = frag_b_f16(Vt, SS, 48, kk);
      o03 = wmma32(pa0, bv3, o03); o13 = wmma32(pa1, bv3, o13);
    }
  }

#pragma unroll
  for (int t = 0; t < 2; ++t) {
    v8f* ot0 = t ? &o10 : &o00; v8f* ot1 = t ? &o11 : &o01;
    v8f* ot2 = t ? &o12 : &o02; v8f* ot3 = t ? &o13 : &o03;
#pragma unroll
    for (int r = 0; r < 8; ++r) {
      const float inv = (lrun[t][r] > 0.f) ? 1.f / lrun[t][r] : 0.f;
      const int s = q0 + t * 16 + r + half8;
      _Float16* dst = o_ws + ((size_t)(b * SS + s) * HH + h) * DHH;
      dst[n]      = (_Float16)((*ot0)[r] * inv);
      dst[16 + n] = (_Float16)((*ot1)[r] * inv);
      dst[32 + n] = (_Float16)((*ot2)[r] * inv);
      dst[48 + n] = (_Float16)((*ot3)[r] * inv);
    }
  }
}

// -------------------------------------------------------- output projection
// 32x64 tile per wave, same register blocking as the QKV GEMM.
__global__ __launch_bounds__(256) void proj_gemm_kernel(
    const _Float16* __restrict__ o_ws, const _Float16* __restrict__ w,
    const float* __restrict__ bias, float* __restrict__ out) {
  const int wave = threadIdx.x >> 5, lane = threadIdx.x & 31;
  const int tile = blockIdx.x * 8 + wave;
  const int m0 = (tile % (M_ROWS / 32)) * 32;
  const int n0 = (tile / (M_ROWS / 32)) * 64;

  v8f acc[2][4] = {};
  for (int k0 = 0; k0 < DD; k0 += 32) {
    const v16h a0 = frag_a_f16(o_ws, DD, m0, k0);
    const v16h a1 = frag_a_f16(o_ws, DD, m0 + 16, k0);
#pragma unroll
    for (int j = 0; j < 4; ++j) {
      const v16h bj = frag_b_f16(w, DD, n0 + j * 16, k0);
      acc[0][j] = wmma32(a0, bj, acc[0][j]);
      acc[1][j] = wmma32(a1, bj, acc[1][j]);
    }
  }

  const int half8 = (lane >> 4) << 3;
#pragma unroll
  for (int j = 0; j < 4; ++j) {
    const int n = n0 + j * 16 + (lane & 15);
    const float bn = bias[n];
#pragma unroll
    for (int i = 0; i < 2; ++i)
#pragma unroll
      for (int r = 0; r < 8; ++r)
        out[(size_t)(m0 + i * 16 + r + half8) * DD + n] = acc[i][j][r] + bn;
  }
}

extern "C" void kernel_launch(void* const* d_in, const int* in_sizes, int n_in,
                              void* d_out, int out_size, void* d_ws,
                              size_t ws_size, hipStream_t stream) {
  (void)in_sizes; (void)n_in; (void)out_size; (void)ws_size;
  const float* query = (const float*)d_in[0];
  const unsigned char* pad = (const unsigned char*)d_in[1];  // bool mask [B,S]
  const float* w_qkv = (const float*)d_in[2];
  const float* b_qkv = (const float*)d_in[3];
  const float* w_out = (const float*)d_in[4];
  const float* b_out = (const float*)d_in[5];
  float* out = (float*)d_out;

  // workspace partition (f16 elements), all 16B-aligned
  const size_t bhsd = (size_t)BB * HH * SS * DHH;  // 3,145,728
  _Float16* base = (_Float16*)d_ws;
  _Float16* q_ws = base;
  _Float16* k_ws = q_ws + bhsd;
  _Float16* v_ws = k_ws + bhsd;                    // transposed [B,H,dh,S]
  _Float16* o_ws = v_ws + bhsd;                    // [B,S,H,dh] == [B,S,D]
  _Float16* xq16 = o_ws + bhsd;                    // query as f16
  _Float16* wq16 = xq16 + (size_t)BB * SS * DD;    // w_qkv as f16
  _Float16* wo16 = wq16 + (size_t)N_QKV * DD;      // w_out as f16

  const int nx4 = BB * SS * DD / 4, nw4 = N_QKV * DD / 4, no4 = DD * DD / 4;
  cvt_f32_to_f16_kernel<<<(nx4 + 255) / 256, 256, 0, stream>>>(query, xq16, nx4);
  cvt_f32_to_f16_kernel<<<(nw4 + 255) / 256, 256, 0, stream>>>(w_qkv, wq16, nw4);
  cvt_f32_to_f16_kernel<<<(no4 + 255) / 256, 256, 0, stream>>>(w_out, wo16, no4);

  qkv_gemm_kernel<<<(M_ROWS / 32) * (N_QKV / 64) / 8, 256, 0, stream>>>(
      xq16, wq16, b_qkv, q_ws, k_ws, v_ws);
  attn_kernel<<<BB * HH * (SS / 32) / 4, 128, 0, stream>>>(q_ws, k_ws, v_ws,
                                                           pad, o_ws);
  proj_gemm_kernel<<<(M_ROWS / 32) * (DD / 64) / 8, 256, 0, stream>>>(
      o_ws, wo16, b_out, out);
}
